// Attention_49933289783881
// MI455X (gfx1250) — matrix operand order repaired and checked by reading, compiled, not run on hardware
//
#include <hip/hip_runtime.h>

typedef __attribute__((ext_vector_type(16))) _Float16 v16h;
typedef __attribute__((ext_vector_type(8)))  float    v8f;
typedef __attribute__((ext_vector_type(4)))  float    v4f;
typedef __attribute__((ext_vector_type(4)))  unsigned v4u;

#define B_    32
#define C_    684
#define H_    32
#define W_    128
#define HID_  256
#define A_    512
#define HW_   4096
#define CPITCH 132      // 132*4B = 528B row pitch: 16B-aligned rows, 4-bank stride for wmma stores
#define WPITCH 136      // weff LDS pitch in halves: 272B rows, 16B aligned
#define ACH    128      // a-channel chunk held in LDS

// ---------------- kernel 1: Weff[a][t] = sum_k Wa[a,k] * Wconv[k,0,t], f16, K padded to 128
__global__ __launch_bounds__(128) void weff_kernel(const float* __restrict__ Wa,
                                                   const float* __restrict__ Wconv,
                                                   _Float16* __restrict__ weff) {
  const int a = blockIdx.x;          // 0..511
  const int t = threadIdx.x;         // 0..127 (taps 0..120 valid)
  float s = 0.f;
  if (t < 121) {
    const float* wa = Wa + (size_t)a * 512;
    for (int k = 0; k < 512; ++k)
      s += wa[k] * Wconv[(size_t)k * 121 + t];   // lanes read consecutive taps: coalesced
  }
  weff[(size_t)a * 128 + t] = (_Float16)s;       // taps >= 121 stay exactly 0
}

// ---------------- kernel 2: qc[b,a] = hidden@Wh.T + counting@Wc.T + bh + bc  (tiny, fp32)
__global__ __launch_bounds__(256) void qc_kernel(const float* __restrict__ hidden,
                                                 const float* __restrict__ counting,
                                                 const float* __restrict__ Wh,
                                                 const float* __restrict__ bh,
                                                 const float* __restrict__ Wc,
                                                 const float* __restrict__ bc,
                                                 float* __restrict__ qc,
                                                 unsigned* __restrict__ maxkey) {
  const int idx = blockIdx.x * 256 + threadIdx.x;  // 0..16383
  if (idx == 0) *maxkey = 0u;                      // re-init ordered max key every launch
  const int b = idx >> 9, a = idx & 511;
  float acc = bh[a] + bc[a];
  const float* hp = hidden   + (size_t)b * HID_;
  const float* cp = counting + (size_t)b * HID_;
  const float* whp = Wh + (size_t)a * HID_;
  const float* wcp = Wc + (size_t)a * HID_;
  for (int k = 0; k < HID_; ++k) acc += hp[k] * whp[k] + cp[k] * wcp[k];
  qc[idx] = acc;
}

// ---------------- kernel 3: fused coverage (WMMA) + tanh + dot(Wv) -> energy[b,h,w]
// One workgroup per (b,h) row: 8 waves, wave q owns positions w = 16q..16q+15.
__global__ __launch_bounds__(256) void attn_energy_kernel(const float* __restrict__ alpha_sum,
                                                          const float* __restrict__ feat,   // [B,A,H,W]
                                                          const float* __restrict__ qc,     // [B,A]
                                                          const _Float16* __restrict__ weff,// [A][128]
                                                          const float* __restrict__ Wv,     // [A]
                                                          const float* __restrict__ bv,
                                                          float* __restrict__ energy) {
  extern __shared__ char smem[];
  float*    lds_alpha = (float*)smem;                                   // [11][144]
  float*    lds_cov   = (float*)(smem + 11 * 144 * 4);                  // [ACH][CPITCH]
  _Float16* lds_weff  = (_Float16*)(smem + 11*144*4 + ACH*CPITCH*4);    // [ACH][WPITCH] halves
  float*    lds_ep    = (float*)(smem + 11*144*4 + ACH*CPITCH*4 + ACH*WPITCH*2); // [1024]

  const int t = threadIdx.x;
  const int b = blockIdx.x >> 5;
  const int h = blockIdx.x & 31;

  // stage 11 haloed alpha_sum rows (global col -5..132) with zero padding
  for (int i = t; i < 11 * 144; i += 256) {
    const int r = i / 144, c = i % 144;
    const int gr = h + r - 5, gc = c - 5;
    float v = 0.f;
    if (gr >= 0 && gr < H_ && gc >= 0 && gc < W_)
      v = alpha_sum[((size_t)b * H_ + gr) * W_ + gc];
    lds_alpha[i] = v;
  }
  __syncthreads();

  const int wave = t >> 5;
  const int lane = t & 31;
  const int m    = lane & 15;   // A-matrix row M held by this lane
  const int hi   = lane >> 4;   // half-wave select
  const int w0   = wave * 16 + m;

  // Build 4 A fragments (16 pos x 128 taps, f16), contiguous k order matching B:
  // element e -> K = 32*ks + 16*hi + e
  v16h afrag[4];
#pragma unroll
  for (int ks = 0; ks < 4; ++ks) {
#pragma unroll
    for (int e = 0; e < 16; ++e) {
      const int K = ks * 32 + hi * 16 + e;
      _Float16 v = (_Float16)0.f;
      if (K < 121) {
        const int dy = K / 11, dx = K % 11;
        v = (_Float16)lds_alpha[dy * 144 + w0 + dx];   // window col = w-5+dx, LDS col0 = -5
      }
      afrag[ks][e] = v;
    }
  }

  const int w4 = t & 31;        // float4 group: w = 4*w4 .. 4*w4+3
  const int ag = t >> 5;        // phase-2 a-group 0..7
  v4f epart = {0.f, 0.f, 0.f, 0.f};

  for (int ch = 0; ch < A_ / ACH; ++ch) {
    const int abase = ch * ACH;

    // ---- stage this chunk's Weff slice (32 KB) into LDS, coalesced b128 copy
    for (int i = t; i < ACH * 16; i += 256) {        // 2048 x 16B
      const int row = i >> 4, seg = i & 15;
      *(v4u*)(lds_weff + row * WPITCH + seg * 8) =
          *(const v4u*)(weff + (size_t)(abase + row) * 128 + seg * 8);
    }
    // ---- prefetch this chunk's feat stream (64 KB) into L2 while WMMAs run
    {
      const int L0 = 2 * t;                           // 512 cachelines of 128B
#pragma unroll
      for (int j = 0; j < 2; ++j) {
        const int r = (L0 + j) >> 2, li = (L0 + j) & 3;
        __builtin_prefetch(feat + (((size_t)b * A_ + abase + r) * H_ + h) * W_ + li * 32, 0, 3);
      }
    }
    __syncthreads();

    // ---- phase 1: coverage via WMMA, B-fragments double-buffered in registers so the
    //      next tile's 8 ds_load_b128 are in flight while the current 4-WMMA chain runs.
    v16h bf[2][4];
#pragma unroll
    for (int ks = 0; ks < 4; ++ks)
      bf[0][ks] = *(const v16h*)(lds_weff + m * WPITCH + ks * 32 + hi * 16);
#pragma unroll
    for (int nt = 0; nt < ACH / 16; ++nt) {
      const int cur = nt & 1, nxt = cur ^ 1;
      if (nt < ACH / 16 - 1) {
#pragma unroll
        for (int ks = 0; ks < 4; ++ks)
          bf[nxt][ks] = *(const v16h*)(lds_weff + ((nt + 1) * 16 + m) * WPITCH + ks * 32 + hi * 16);
      }
      v8f acc = {};
#pragma unroll
      for (int ks = 0; ks < 4; ++ks)
        acc = __builtin_amdgcn_wmma_f32_16x16x32_f16(false, afrag[ks], false, bf[cur][ks],
                                                     (short)0, acc, false, false);
#pragma unroll
      for (int r = 0; r < 8; ++r) {
        const int al = nt * 16 + m;             // N (a within chunk)
        const int wp = wave * 16 + r + hi * 8;  // M (position)
        lds_cov[al * CPITCH + wp] = acc[r];
      }
    }
    __syncthreads();

    // ---- phase 2: score = tanh(cov + qc + feat); energy partial dot with Wv (b128 loads)
#pragma unroll
    for (int i = 0; i < 16; ++i) {
      const int al = i * 8 + ag;
      const int a  = abase + al;
      const v4f cov = *(const v4f*)(lds_cov + al * CPITCH + w4 * 4);
      const v4f f   = *(const v4f*)(feat + (((size_t)b * A_ + a) * H_ + h) * W_ + w4 * 4);
      const float q  = qc[(size_t)b * A_ + a];
      const float wv = Wv[a];
      v4f s;
      s.x = tanhf(cov.x + q + f.x);
      s.y = tanhf(cov.y + q + f.y);
      s.z = tanhf(cov.z + q + f.z);
      s.w = tanhf(cov.w + q + f.w);
      epart += s * wv;
    }
    __syncthreads();
  }

  // reduce 8 a-groups per w; thread t wrote components for w = 4*w4..4*w4+3
#pragma unroll
  for (int cmp = 0; cmp < 4; ++cmp) lds_ep[t * 4 + cmp] = epart[cmp];
  __syncthreads();
  if (t < 128) {
    float e = bv[0];
#pragma unroll
    for (int g = 0; g < 8; ++g) e += lds_ep[g * 128 + t];
    energy[((size_t)b * H_ + h) * W_ + t] = e;
  }
}

// ---------------- kernel 4: global max via order-preserving uint key
__device__ __forceinline__ unsigned fkey(float f) {
  unsigned u = __float_as_uint(f);
  return (u & 0x80000000u) ? ~u : (u | 0x80000000u);
}
__global__ __launch_bounds__(256) void max_kernel(const float* __restrict__ energy,
                                                  unsigned* __restrict__ maxkey) {
  __shared__ float red[256];
  const int t = threadIdx.x;
  float mv = -3.4e38f;
  for (int i = 0; i < 4; ++i) mv = fmaxf(mv, energy[blockIdx.x * 1024 + i * 256 + t]);
  red[t] = mv;
  __syncthreads();
  for (int s = 128; s > 0; s >>= 1) {
    if (t < s) red[t] = fmaxf(red[t], red[t + s]);
    __syncthreads();
  }
  if (t == 0) atomicMax(maxkey, fkey(red[0]));
}

// ---------------- kernel 5: per-batch softmax (+1e-10), alpha & alpha_sum_new
__global__ __launch_bounds__(256) void softmax_kernel(const float* __restrict__ energy,
                                                      const unsigned* __restrict__ maxkey,
                                                      const float* __restrict__ mask,
                                                      const float* __restrict__ alpha_sum_in,
                                                      float* __restrict__ out_alpha,
                                                      float* __restrict__ out_asum) {
  __shared__ float lds_e[HW_];
  __shared__ float red[256];
  const int b = blockIdx.x, t = threadIdx.x;
  const unsigned k = *maxkey;
  const float gmax = __uint_as_float((k & 0x80000000u) ? (k & 0x7fffffffu) : ~k);
  float psum = 0.f;
  for (int i = 0; i < 16; ++i) {
    const int idx = i * 256 + t;
    const float v = expf(energy[(size_t)b * HW_ + idx] - gmax) * mask[(size_t)b * HW_ + idx];
    lds_e[idx] = v;
    psum += v;
  }
  red[t] = psum;
  __syncthreads();
  for (int s = 128; s > 0; s >>= 1) {
    if (t < s) red[t] += red[t + s];
    __syncthreads();
  }
  const float inv = 1.f / (red[0] + 1e-10f);
  for (int i = 0; i < 16; ++i) {
    const int idx = i * 256 + t;
    const float a = lds_e[idx] * inv;
    out_alpha[(size_t)b * HW_ + idx] = a;
    out_asum[(size_t)b * HW_ + idx]  = a + alpha_sum_in[(size_t)b * HW_ + idx];
  }
}

// ---------------- kernel 6: context[b,c] = dot(cnn[b,c,:], alpha[b,:])  (bandwidth bound)
// 358 MB stream > 192 MB L2 -> nontemporal b128 loads; alpha (16 KB/b) stays cached.
__global__ __launch_bounds__(256) void context_kernel(const float* __restrict__ cnn,
                                                      const float* __restrict__ alpha,
                                                      float* __restrict__ ctx) {
  const int b = blockIdx.x / 86, g = blockIdx.x % 86;
  const int wave = threadIdx.x >> 5, lane = threadIdx.x & 31;
  const int c = g * 8 + wave;
  v4f acc = {0.f, 0.f, 0.f, 0.f};
  if (c < C_) {
    const v4f* cp = (const v4f*)(cnn + ((size_t)b * C_ + c) * HW_);
    const v4f* ap = (const v4f*)(alpha + (size_t)b * HW_);
#pragma unroll 4
    for (int i = 0; i < 32; ++i) {
      const int idx = i * 32 + lane;             // coalesced b128
      const v4f x = __builtin_nontemporal_load(cp + idx);
      const v4f al = ap[idx];
      acc += x * al;
    }
  }
  float s = acc.x + acc.y + acc.z + acc.w;
  for (int off = 16; off > 0; off >>= 1) s += __shfl_xor(s, off, 32);
  if (c < C_ && lane == 0) ctx[(size_t)b * C_ + c] = s;
}

extern "C" void kernel_launch(void* const* d_in, const int* in_sizes, int n_in,
                              void* d_out, int out_size, void* d_ws, size_t ws_size,
                              hipStream_t stream) {
  (void)in_sizes; (void)n_in; (void)out_size; (void)ws_size;
  const float* cnn   = (const float*)d_in[0];
  const float* featT = (const float*)d_in[1];
  const float* hidden= (const float*)d_in[2];
  const float* asum  = (const float*)d_in[3];
  const float* cdyn  = (const float*)d_in[4];
  const float* mask  = (const float*)d_in[5];
  const float* Wh    = (const float*)d_in[6];
  const float* bh    = (const float*)d_in[7];
  const float* Wc    = (const float*)d_in[8];
  const float* bc    = (const float*)d_in[9];
  const float* Wconv = (const float*)d_in[10];
  const float* Wa    = (const float*)d_in[11];
  const float* Wv    = (const float*)d_in[12];
  const float* bv    = (const float*)d_in[13];

  char* ws = (char*)d_ws;
  float*    ws_qc     = (float*)ws;                                  // 64 KB
  _Float16* ws_weff   = (_Float16*)(ws + 65536);                     // 128 KB
  float*    ws_energy = (float*)(ws + 65536 + 131072);               // 512 KB
  unsigned* ws_key    = (unsigned*)(ws + 65536 + 131072 + 524288);   // 4 B

  float* out_ctx   = (float*)d_out;               // [B,C]   = 21888
  float* out_alpha = out_ctx + B_ * C_;           // [B,H,W] = 131072
  float* out_asum  = out_alpha + B_ * HW_;        // [B,1,H,W]

  weff_kernel<<<A_, 128, 0, stream>>>(Wa, Wconv, ws_weff);
  qc_kernel<<<64, 256, 0, stream>>>(hidden, cdyn, Wh, bh, Wc, bc, ws_qc, ws_key);
  const size_t shb = 11*144*4 + ACH*CPITCH*4 + ACH*WPITCH*2 + 1024*4;   // ~110 KB
  attn_energy_kernel<<<B_ * H_, 256, shb, stream>>>(asum, featT, ws_qc, ws_weff, Wv, bv, ws_energy);
  max_kernel<<<128, 256, 0, stream>>>(ws_energy, ws_key);
  softmax_kernel<<<B_, 256, 0, stream>>>(ws_energy, ws_key, mask, asum, out_alpha, out_asum);
  context_kernel<<<B_ * 86, 256, 0, stream>>>(cnn, out_alpha, out_ctx);
}
